// FeedBack_21088289423626
// MI455X (gfx1250) — matrix operand order repaired and checked by reading, not compiled, not run on hardware
//
#include <hip/hip_runtime.h>
#include <hip/hip_bf16.h>
#include <math.h>

// ---------------------------------------------------------------------------
// Problem constants (from reference): B=1024, T=100, F=128, H=512, OUT=60
// ---------------------------------------------------------------------------
#define B_SZ      1024
#define T_SEQ     100
#define F_DIM     128
#define H_DIM     512
#define OUT_STEPS 60
#define GATES     (4 * H_DIM)          // 2048
#define BROWS     16                    // batch rows per block (WMMA M)
#define NBLOCKS   (B_SZ / BROWS)        // 64 persistent blocks
#define NTHREADS  256                   // 8 waves of 32

typedef __attribute__((ext_vector_type(16))) __bf16 v16bf;
typedef __attribute__((ext_vector_type(8)))  __bf16 v8bf;
typedef __attribute__((ext_vector_type(8)))  float  v8f;

#define GLOBAL_AS __attribute__((address_space(1)))

// ---------------------------------------------------------------------------
// WMMA helper: D = A(16x32 bf16) * B(32x16 bf16) + C(16x16 f32)
// ---------------------------------------------------------------------------
__device__ __forceinline__ v8f wmma_bf16(v16bf a, v16bf b, v8f c) {
    return __builtin_amdgcn_wmma_f32_16x16x32_bf16(
        /*neg_a=*/false, a, /*neg_b=*/false, b,
        /*c_mod=*/(short)0, c, /*reuse_a=*/false, /*reuse_b=*/false);
}

// A-fragment (16x32, row-major bf16 in LDS, row stride = `stride` elements).
// ISA layout: lanes 0-15 hold row M=lane, K = {k0..k0+7, k0+16..k0+23};
//             lanes 16-31 hold row M=lane-16, K = {k0+8..k0+15, k0+24..k0+31}.
__device__ __forceinline__ v16bf lds_load_A(const __bf16* s, int stride, int k0) {
    const int lane = threadIdx.x & 31;
    const int m    = lane & 15;
    const int kh   = (lane >> 4) << 3;          // 0 or 8
    const __bf16* p = s + m * stride + k0 + kh;
    v8bf lo = *(const v8bf*)(p);                // 16B -> ds_load_b128
    v8bf hi = *(const v8bf*)(p + 16);
    return __builtin_shufflevector(lo, hi, 0,1,2,3,4,5,6,7,8,9,10,11,12,13,14,15);
}

// B-fragment (32x16) from transposed bf16 weights WT[N][K] in global memory.
// Explicit address_space(1) so the backend emits global_load_b128 (not flat):
// flat loads would count against both LOADcnt and DScnt and serialize with LDS.
// ISA layout: lanes 0-15 hold col N=lane, K = {k0..k0+7, k0+16..k0+23};
//             lanes 16-31 K = {k0+8..k0+15, k0+24..k0+31}.
__device__ __forceinline__ v16bf glb_load_B(const __bf16* wt,
                                            int K, int ncol_base, int k0) {
    const int lane = threadIdx.x & 31;
    const int n    = ncol_base + (lane & 15);
    const int ko   = k0 + ((lane >> 4) << 3);   // 0 or 8
    const GLOBAL_AS __bf16* p =
        (const GLOBAL_AS __bf16*)(wt + (size_t)n * K + ko);
    v8bf lo = *(const GLOBAL_AS v8bf*)(p);      // 16B -> global_load_b128
    v8bf hi = *(const GLOBAL_AS v8bf*)(p + 16);
    return __builtin_shufflevector(lo, hi, 0,1,2,3,4,5,6,7,8,9,10,11,12,13,14,15);
}

__device__ __forceinline__ float sigmoidf_(float x) {
    return 1.0f / (1.0f + __expf(-x));
}

// ---------------------------------------------------------------------------
// Prep: convert f32 src[K][N] (row-major) -> bf16 dst[N][K] (transposed)
// ---------------------------------------------------------------------------
__global__ void wconv_kernel(const float* __restrict__ src,
                             __bf16* __restrict__ dst, int K, int N) {
    int idx = blockIdx.x * blockDim.x + threadIdx.x;
    if (idx >= K * N) return;
    int k = idx / N;
    int n = idx - k * N;
    dst[(size_t)n * K + k] = (__bf16)src[idx];
}

// ---------------------------------------------------------------------------
// Persistent 2-layer LSTM + feedback decoder. One block = 16 batch rows,
// runs all 159 time steps independently (recurrence is per-batch-row).
// Wave w owns hidden units [w*64, w*64+64); processed in 4 passes of 16
// columns so only the 4 gate tiles (i,f,g,o) are live at once. Cell state
// c1/c2 stays in registers for the whole run. h is double-buffered in LDS
// (ping-pong on t&1) so gate results stream straight to the write buffer.
// ---------------------------------------------------------------------------
__global__ __launch_bounds__(NTHREADS, 1)
void lstm_feedback_kernel(const float* __restrict__ inputs,   // [B,T,F] f32
                          const float* __restrict__ g_b1,     // [4H]
                          const float* __restrict__ g_b2,     // [4H]
                          const float* __restrict__ g_bd,     // [F]
                          const __bf16* __restrict__ g_WxT1,  // [4H][F]
                          const __bf16* __restrict__ g_WhT1,  // [4H][H]
                          const __bf16* __restrict__ g_WxT2,  // [4H][H]
                          const __bf16* __restrict__ g_WhT2,  // [4H][H]
                          const __bf16* __restrict__ g_WdT,   // [F][H]
                          float* __restrict__ out)            // [B,OUT,F] f32
{
    __shared__ __align__(16) __bf16 sX [BROWS * F_DIM];       //  4 KB
    __shared__ __align__(16) __bf16 sH1[2][BROWS * H_DIM];    // 32 KB
    __shared__ __align__(16) __bf16 sH2[2][BROWS * H_DIM];    // 32 KB

    const int tid   = threadIdx.x;
    const int lane  = tid & 31;
    const int wave  = tid >> 5;            // 0..7
    const int blk   = blockIdx.x;          // 0..63
    const int hbase = wave * 64;           // this wave's hidden-unit slice
    const int ln15  = lane & 15;
    const int mhalf = (lane >> 4) << 3;    // 0 or 8: C/D tile row base

    // h1 = h2 = 0 in the first read buffer
    for (int i = tid; i < BROWS * H_DIM; i += NTHREADS) {
        sH1[0][i] = (__bf16)0.0f;
        sH2[0][i] = (__bf16)0.0f;
    }

    // Cell states live in registers for the whole run: 16x64 per layer/wave.
    v8f c1[4], c2[4];
    #pragma unroll
    for (int t4 = 0; t4 < 4; ++t4)
        #pragma unroll
        for (int e = 0; e < 8; ++e) { c1[t4][e] = 0.0f; c2[t4][e] = 0.0f; }

    for (int t = 0; t < T_SEQ + OUT_STEPS - 1; ++t) {       // 159 steps
        const int rb = t & 1;              // read buffer
        const int wb = rb ^ 1;             // write buffer
        const __bf16* h1r = sH1[rb];
        __bf16*       h1w = sH1[wb];
        const __bf16* h2r = sH2[rb];
        __bf16*       h2w = sH2[wb];

        // ---- stage x_t (warm phase) : f32 -> bf16 into sX -------------
        if (t < T_SEQ) {
            const int i0 = tid * 8;                 // 2048 elems / 256 thr
            const int m  = i0 >> 7;                 // /F_DIM
            const int f  = i0 & (F_DIM - 1);
            const float* src = inputs +
                ((size_t)(blk * BROWS + m) * T_SEQ + t) * F_DIM + f;
            const float4 xa = ((const float4*)src)[0];
            const float4 xb = ((const float4*)src)[1];
            __bf16* dst = &sX[i0];
            dst[0] = (__bf16)xa.x; dst[1] = (__bf16)xa.y;
            dst[2] = (__bf16)xa.z; dst[3] = (__bf16)xa.w;
            dst[4] = (__bf16)xb.x; dst[5] = (__bf16)xb.y;
            dst[6] = (__bf16)xb.z; dst[7] = (__bf16)xb.w;
        }
        __syncthreads();

        // Re-materialize weight/bias pointers through an opaque asm barrier:
        // prevents the compiler from hoisting the (t-invariant) weight loads
        // out of the time loop and spilling hundreds of VGPRs to scratch.
        const __bf16* WxT1 = g_WxT1;  const __bf16* WhT1 = g_WhT1;
        const __bf16* WxT2 = g_WxT2;  const __bf16* WhT2 = g_WhT2;
        const __bf16* WdT  = g_WdT;
        const float*  b1   = g_b1;    const float*  b2   = g_b2;
        const float*  bd   = g_bd;
        asm volatile("" : "+s"(WxT1), "+s"(WhT1), "+s"(WxT2), "+s"(WhT2),
                          "+s"(WdT), "+s"(b1), "+s"(b2), "+s"(bd));

        // =========================== layer 1 ===========================
        #pragma unroll
        for (int t4 = 0; t4 < 4; ++t4) {
            const int nb = hbase + t4 * 16;         // tile column base
            const int cc = nb + ln15;               // this lane's column
            v8f zi = {0,0,0,0,0,0,0,0}, zf = {0,0,0,0,0,0,0,0};
            v8f zg = {0,0,0,0,0,0,0,0}, zo = {0,0,0,0,0,0,0,0};
            for (int k0 = 0; k0 < F_DIM; k0 += 32) {            // x @ Wx1
                const v16bf a = lds_load_A(sX, F_DIM, k0);
                zi = wmma_bf16(a, glb_load_B(WxT1, F_DIM, 0 * H_DIM + nb, k0), zi);
                zf = wmma_bf16(a, glb_load_B(WxT1, F_DIM, 1 * H_DIM + nb, k0), zf);
                zg = wmma_bf16(a, glb_load_B(WxT1, F_DIM, 2 * H_DIM + nb, k0), zg);
                zo = wmma_bf16(a, glb_load_B(WxT1, F_DIM, 3 * H_DIM + nb, k0), zo);
            }
            for (int k0 = 0; k0 < H_DIM; k0 += 32) {            // h1 @ Wh1
                const v16bf a = lds_load_A(h1r, H_DIM, k0);
                zi = wmma_bf16(a, glb_load_B(WhT1, H_DIM, 0 * H_DIM + nb, k0), zi);
                zf = wmma_bf16(a, glb_load_B(WhT1, H_DIM, 1 * H_DIM + nb, k0), zf);
                zg = wmma_bf16(a, glb_load_B(WhT1, H_DIM, 2 * H_DIM + nb, k0), zg);
                zo = wmma_bf16(a, glb_load_B(WhT1, H_DIM, 3 * H_DIM + nb, k0), zo);
            }
            const float bi = b1[0 * H_DIM + cc];
            const float bf = b1[1 * H_DIM + cc];
            const float bg = b1[2 * H_DIM + cc];
            const float bo = b1[3 * H_DIM + cc];
            #pragma unroll
            for (int e = 0; e < 8; ++e) {                       // gates
                const float cn = sigmoidf_(zf[e] + bf) * c1[t4][e]
                               + sigmoidf_(zi[e] + bi) * tanhf(zg[e] + bg);
                c1[t4][e] = cn;
                h1w[(mhalf + e) * H_DIM + cc] =
                    (__bf16)(sigmoidf_(zo[e] + bo) * tanhf(cn));
            }
        }
        __syncthreads();        // h1_new (write buffer) complete & visible

        // =========================== layer 2 ===========================
        #pragma unroll
        for (int t4 = 0; t4 < 4; ++t4) {
            const int nb = hbase + t4 * 16;
            const int cc = nb + ln15;
            v8f zi = {0,0,0,0,0,0,0,0}, zf = {0,0,0,0,0,0,0,0};
            v8f zg = {0,0,0,0,0,0,0,0}, zo = {0,0,0,0,0,0,0,0};
            for (int k0 = 0; k0 < H_DIM; k0 += 32) {            // h1 @ Wx2
                const v16bf a = lds_load_A(h1w, H_DIM, k0);
                zi = wmma_bf16(a, glb_load_B(WxT2, H_DIM, 0 * H_DIM + nb, k0), zi);
                zf = wmma_bf16(a, glb_load_B(WxT2, H_DIM, 1 * H_DIM + nb, k0), zf);
                zg = wmma_bf16(a, glb_load_B(WxT2, H_DIM, 2 * H_DIM + nb, k0), zg);
                zo = wmma_bf16(a, glb_load_B(WxT2, H_DIM, 3 * H_DIM + nb, k0), zo);
            }
            for (int k0 = 0; k0 < H_DIM; k0 += 32) {            // h2 @ Wh2
                const v16bf a = lds_load_A(h2r, H_DIM, k0);
                zi = wmma_bf16(a, glb_load_B(WhT2, H_DIM, 0 * H_DIM + nb, k0), zi);
                zf = wmma_bf16(a, glb_load_B(WhT2, H_DIM, 1 * H_DIM + nb, k0), zf);
                zg = wmma_bf16(a, glb_load_B(WhT2, H_DIM, 2 * H_DIM + nb, k0), zg);
                zo = wmma_bf16(a, glb_load_B(WhT2, H_DIM, 3 * H_DIM + nb, k0), zo);
            }
            const float bi = b2[0 * H_DIM + cc];
            const float bf = b2[1 * H_DIM + cc];
            const float bg = b2[2 * H_DIM + cc];
            const float bo = b2[3 * H_DIM + cc];
            #pragma unroll
            for (int e = 0; e < 8; ++e) {
                const float cn = sigmoidf_(zf[e] + bf) * c2[t4][e]
                               + sigmoidf_(zi[e] + bi) * tanhf(zg[e] + bg);
                c2[t4][e] = cn;
                h2w[(mhalf + e) * H_DIM + cc] =
                    (__bf16)(sigmoidf_(zo[e] + bo) * tanhf(cn));
            }
        }
        __syncthreads();        // h2_new complete & visible

        // ===================== decoder (pred = h2 @ Wd + bd) ===========
        if (t >= T_SEQ - 1) {
            const int s = t - (T_SEQ - 1);                  // 0..59
            v8f p = {0,0,0,0,0,0,0,0};
            for (int k0 = 0; k0 < H_DIM; k0 += 32) {
                const v16bf a = lds_load_A(h2w, H_DIM, k0);
                p = wmma_bf16(a, glb_load_B(WdT, H_DIM, wave * 16, k0), p);
            }
            const int n  = wave * 16 + ln15;                // output feature
            const float bv = bd[n];
            #pragma unroll
            for (int e = 0; e < 8; ++e) {
                const int m  = mhalf + e;
                const float pv = p[e] + bv;
                out[((size_t)(blk * BROWS + m) * OUT_STEPS + s) * F_DIM + n] = pv;
                sX[m * F_DIM + n] = (__bf16)pv;             // feedback input
            }
        }
        __syncthreads();        // sX stable; read buffers free for reuse
    }
}

// ---------------------------------------------------------------------------
// Host launcher
// ---------------------------------------------------------------------------
extern "C" void kernel_launch(void* const* d_in, const int* in_sizes, int n_in,
                              void* d_out, int out_size, void* d_ws, size_t ws_size,
                              hipStream_t stream) {
    (void)in_sizes; (void)n_in; (void)out_size; (void)ws_size;

    const float* inputs = (const float*)d_in[0];   // [B,T,F]
    const float* Wx1    = (const float*)d_in[1];   // [F,4H]
    const float* Wh1    = (const float*)d_in[2];   // [H,4H]
    const float* b1     = (const float*)d_in[3];   // [4H]
    const float* Wx2    = (const float*)d_in[4];   // [H,4H]
    const float* Wh2    = (const float*)d_in[5];   // [H,4H]
    const float* b2     = (const float*)d_in[6];   // [4H]
    const float* Wd     = (const float*)d_in[7];   // [H,F]
    const float* bd     = (const float*)d_in[8];   // [F]

    // Workspace: transposed bf16 weights (~6.9 MB total)
    __bf16* WxT1 = (__bf16*)d_ws;                         // [2048][128]
    __bf16* WhT1 = WxT1 + (size_t)GATES * F_DIM;          // [2048][512]
    __bf16* WxT2 = WhT1 + (size_t)GATES * H_DIM;          // [2048][512]
    __bf16* WhT2 = WxT2 + (size_t)GATES * H_DIM;          // [2048][512]
    __bf16* WdT  = WhT2 + (size_t)GATES * H_DIM;          // [128][512]

    const int thr = 256;
    wconv_kernel<<<(F_DIM * GATES + thr - 1) / thr, thr, 0, stream>>>(Wx1, WxT1, F_DIM, GATES);
    wconv_kernel<<<(H_DIM * GATES + thr - 1) / thr, thr, 0, stream>>>(Wh1, WhT1, H_DIM, GATES);
    wconv_kernel<<<(H_DIM * GATES + thr - 1) / thr, thr, 0, stream>>>(Wh2 ? Wx2 : Wx2, WxT2, H_DIM, GATES);
    wconv_kernel<<<(H_DIM * GATES + thr - 1) / thr, thr, 0, stream>>>(Wh2, WhT2, H_DIM, GATES);
    wconv_kernel<<<(H_DIM * F_DIM + thr - 1) / thr, thr, 0, stream>>>(Wd,  WdT,  H_DIM, F_DIM);

    lstm_feedback_kernel<<<NBLOCKS, NTHREADS, 0, stream>>>(
        inputs, b1, b2, bd, WxT1, WhT1, WxT2, WhT2, WdT, (float*)d_out);
}
